// CorrelationNet_86320252715513
// MI455X (gfx1250) — compile-verified, matrix operand order checked by reading, not run on hardware
//
#include <hip/hip_runtime.h>

// ---------------- problem constants ----------------
#define B_  8
#define C_  96
#define H_  160
#define W_  320
#define HP  162          // H + 2 halo
#define WP  322          // W + 2 halo
#define NPL 128          // PLANES
#define CORRK 96         // 81 correlation channels padded to 96 (K mult of 32)
#define NCORR 81

// ---------------- vector types for WMMA ----------------
typedef __attribute__((ext_vector_type(16))) __bf16        v16bf;
typedef __attribute__((ext_vector_type(8)))  float         v8f;
typedef __attribute__((ext_vector_type(16))) unsigned short u16x16;
typedef __attribute__((ext_vector_type(4)))  unsigned int   u32x4;

union Frag { v16bf f; u16x16 s; u32x4 q[2]; };

__device__ __forceinline__ void loadFrag(Frag& f, const unsigned short* base, int off) {
    f.q[0] = *(const u32x4*)(base + off);
    f.q[1] = *(const u32x4*)(base + off + 16);
}

__device__ __forceinline__ unsigned short f2bf(float x) {
    unsigned int u = __float_as_uint(x);
    u += 0x7FFFu + ((u >> 16) & 1u);          // round-to-nearest-even
    return (unsigned short)(u >> 16);
}
__device__ __forceinline__ float bf2f(unsigned short h) {
    return __uint_as_float(((unsigned int)h) << 16);
}

// ---------------- small prep kernels ----------------
__global__ void zero_stats_kernel(float* stats) {
    int i = threadIdx.x;                       // 256 threads, 768 floats
    for (; i < 3 * 2 * NPL; i += 256) stats[i] = 0.f;
}

__global__ void pack_w1_kernel(const float* __restrict__ w, unsigned short* __restrict__ out, int n) {
    int idx = blockIdx.x * blockDim.x + threadIdx.x;
    if (idx >= n) return;                      // layout out[tap][n][k96]
    int k = idx % CORRK, oc = (idx / CORRK) % NPL, tap = idx / (CORRK * NPL);
    int ky = tap / 3, kx = tap % 3;
    float v = (k < NCORR) ? w[((oc * NCORR + k) * 3 + ky) * 3 + kx] : 0.f;
    out[idx] = f2bf(v);
}

__global__ void pack_w2_kernel(const float* __restrict__ w, unsigned short* __restrict__ out, int n) {
    int idx = blockIdx.x * blockDim.x + threadIdx.x;
    if (idx >= n) return;                      // layout out[tap][n][k128]
    int k = idx % NPL, oc = (idx / NPL) % NPL, tap = idx / (NPL * NPL);
    int ky = tap / 3, kx = tap % 3;
    out[idx] = f2bf(w[((oc * NPL + k) * 3 + ky) * 3 + kx]);
}

__global__ void pack_wds_kernel(const float* __restrict__ w, unsigned short* __restrict__ out, int n) {
    int idx = blockIdx.x * blockDim.x + threadIdx.x;
    if (idx >= n) return;                      // layout out[n][k96]
    int k = idx % CORRK, oc = idx / CORRK;
    float v = (k < NCORR) ? w[oc * NCORR + k] : 0.f;
    out[idx] = f2bf(v);
}

__global__ void halo_zero_kernel(unsigned short* __restrict__ corr,
                                 unsigned short* __restrict__ act1, int n) {
    int idx = blockIdx.x * blockDim.x + threadIdx.x;
    if (idx >= n) return;                      // n = B*HP*WP padded pixels
    int p = idx % (HP * WP);
    int hp = p / WP, wp = p % WP;
    if (hp == 0 || hp == HP - 1 || wp == 0 || wp == WP - 1) {
        unsigned short* c = corr + (size_t)idx * CORRK;
        for (int i = 0; i < CORRK; ++i) c[i] = 0;
        unsigned short* a = act1 + (size_t)idx * NPL;
        for (int i = 0; i < NPL; ++i) a[i] = 0;
    }
}

// ---------------- correlation volume ----------------
// grid (W/64, H, B), block 64.  out: channel-last padded bf16 [B][HP][WP][96]
__global__ void corr_kernel(const float* __restrict__ f1, const float* __restrict__ f2,
                            unsigned short* __restrict__ corr) {
    __shared__ float s[9][72];
    const int tid = threadIdx.x;
    const int b = blockIdx.z, h = blockIdx.y, w0 = blockIdx.x * 64;
    const int w = w0 + tid;

    float acc[NCORR];
#pragma unroll
    for (int d = 0; d < NCORR; ++d) acc[d] = 0.f;

    for (int c = 0; c < C_; ++c) {
        const float f1v = f1[(((size_t)b * C_ + c) * H_ + h) * W_ + w];
        for (int i = tid; i < 9 * 72; i += 64) {
            int ry = i / 72, rx = i % 72;
            int hh = h - 4 + ry, ww = w0 - 4 + rx;
            float v = 0.f;
            if (hh >= 0 && hh < H_ && ww >= 0 && ww < W_)
                v = f2[(((size_t)b * C_ + c) * H_ + hh) * W_ + ww];
            s[ry][rx] = v;
        }
        __syncthreads();
#pragma unroll
        for (int dy = 0; dy < 9; ++dy)
#pragma unroll
            for (int dx = 0; dx < 9; ++dx)
                acc[dy * 9 + dx] = fmaf(f1v, s[dy][tid + dx], acc[dy * 9 + dx]);
        __syncthreads();
    }

    unsigned short* cp = corr + (((size_t)b * HP + (h + 1)) * WP + (w + 1)) * CORRK;
    const float inv = 1.f / (float)C_;
#pragma unroll
    for (int d = 0; d < NCORR; ++d) cp[d] = f2bf(acc[d] * inv);
#pragma unroll
    for (int d = NCORR; d < CORRK; ++d) cp[d] = 0;
}

// ---------------- step offset helpers (compile-time after unroll) ----------------
__device__ __forceinline__ constexpr int a_off_c1(int s) {   // conv1: 27 steps
    int tap = s / 3, kc = (s % 3) * 32;
    return ((tap / 3) * WP + (tap % 3)) * CORRK + kc;
}
__device__ __forceinline__ constexpr int b_off_c1(int s, int t) {
    int tap = s / 3, kc = (s % 3) * 32;
    return (tap * NPL + t * 16) * CORRK + kc;
}
__device__ __forceinline__ constexpr int a_off_c2(int s) {   // conv2: 36 steps
    int tap = s / 4, kc = (s % 4) * 32;
    return ((tap / 3) * WP + (tap % 3)) * NPL + kc;
}
__device__ __forceinline__ constexpr int b_off_c2(int s, int t) {
    int tap = s / 4, kc = (s % 4) * 32;
    return (tap * NPL + t * 16) * NPL + kc;
}

// ---------------- generic stats flush helper ----------------
__device__ __forceinline__ void flush_stats(float* sSum, float* sSq, float* stats) {
    __syncthreads();
    if (threadIdx.x < NPL) {
        atomicAdd(&stats[threadIdx.x], sSum[threadIdx.x]);
        atomicAdd(&stats[NPL + threadIdx.x], sSq[threadIdx.x]);
    }
}

// ---------------- conv1: 3x3, 81(->96)x9 -> 128, implicit GEMM via WMMA ----------------
// grid (W/64, H, B), block 256 (8 waves). act1 out: bf16 [B][HP][WP][128] (+halo).
__global__ void conv1_gemm_kernel(const unsigned short* __restrict__ corr,
                                  const unsigned short* __restrict__ w1,
                                  unsigned short* __restrict__ act1,
                                  float* __restrict__ stats) {
    __shared__ float sSum[NPL], sSq[NPL];
    if (threadIdx.x < NPL) { sSum[threadIdx.x] = 0.f; sSq[threadIdx.x] = 0.f; }
    __syncthreads();

    const int b = blockIdx.z, h = blockIdx.y, w0 = blockIdx.x * 64;
    const int wave = threadIdx.x >> 5, lane = threadIdx.x & 31;
    const int mw = wave & 3, nw = wave >> 2;
    const int m0 = mw * 16, n0 = nw * 64;
    const int l15 = lane & 15, kb = (lane >> 4) * 8;

    const unsigned short* abase =
        corr + (((size_t)b * HP + h) * WP + (w0 + m0 + l15)) * CORRK + kb;
    const unsigned short* bbase = w1 + (size_t)(n0 + l15) * CORRK + kb;

    v8f acc[4] = {};
    constexpr int NS = 27;

    // software-pipelined: next step's fragments load while current step's WMMAs run
    Frag aC, bC0, bC1, bC2, bC3;
    loadFrag(aC, abase, a_off_c1(0));
    loadFrag(bC0, bbase, b_off_c1(0, 0));
    loadFrag(bC1, bbase, b_off_c1(0, 1));
    loadFrag(bC2, bbase, b_off_c1(0, 2));
    loadFrag(bC3, bbase, b_off_c1(0, 3));
#pragma unroll
    for (int s = 0; s < NS; ++s) {
        Frag aN, bN0, bN1, bN2, bN3;
        if (s + 1 < NS) {
            loadFrag(aN, abase, a_off_c1(s + 1));
            loadFrag(bN0, bbase, b_off_c1(s + 1, 0));
            loadFrag(bN1, bbase, b_off_c1(s + 1, 1));
            loadFrag(bN2, bbase, b_off_c1(s + 1, 2));
            loadFrag(bN3, bbase, b_off_c1(s + 1, 3));
        }
        acc[0] = __builtin_amdgcn_wmma_f32_16x16x32_bf16(false, aC.f, false, bC0.f, (short)0, acc[0], false, false);
        acc[1] = __builtin_amdgcn_wmma_f32_16x16x32_bf16(false, aC.f, false, bC1.f, (short)0, acc[1], false, false);
        acc[2] = __builtin_amdgcn_wmma_f32_16x16x32_bf16(false, aC.f, false, bC2.f, (short)0, acc[2], false, false);
        acc[3] = __builtin_amdgcn_wmma_f32_16x16x32_bf16(false, aC.f, false, bC3.f, (short)0, acc[3], false, false);
        aC = aN; bC0 = bN0; bC1 = bN1; bC2 = bN2; bC3 = bN3;
    }

    const int mbase = m0 + ((lane >> 4) * 8);
    unsigned short* sbase =
        act1 + (((size_t)b * HP + (h + 1)) * WP + (w0 + mbase + 1)) * NPL + (n0 + l15);
#pragma unroll
    for (int t = 0; t < 4; ++t) {
        float lsum = 0.f, lsq = 0.f;
#pragma unroll
        for (int g = 0; g < 8; ++g) {
            float x = acc[t][g];
            sbase[g * NPL + t * 16] = f2bf(x);
            lsum += x; lsq += x * x;
        }
        atomicAdd(&sSum[n0 + t * 16 + l15], lsum);
        atomicAdd(&sSq[n0 + t * 16 + l15], lsq);
    }
    flush_stats(sSum, sSq, stats);
}

// ---------------- 1x1 downsample conv: 81(->96) -> 128, WMMA ----------------
__global__ void ds_gemm_kernel(const unsigned short* __restrict__ corr,
                               const unsigned short* __restrict__ wds,
                               unsigned short* __restrict__ res,
                               float* __restrict__ stats) {
    __shared__ float sSum[NPL], sSq[NPL];
    if (threadIdx.x < NPL) { sSum[threadIdx.x] = 0.f; sSq[threadIdx.x] = 0.f; }
    __syncthreads();

    const int b = blockIdx.z, h = blockIdx.y, w0 = blockIdx.x * 64;
    const int wave = threadIdx.x >> 5, lane = threadIdx.x & 31;
    const int mw = wave & 3, nw = wave >> 2;
    const int m0 = mw * 16, n0 = nw * 64;
    const int l15 = lane & 15, kb = (lane >> 4) * 8;

    const unsigned short* abase =
        corr + (((size_t)b * HP + (h + 1)) * WP + (w0 + m0 + l15 + 1)) * CORRK + kb;
    const unsigned short* bbase = wds + (size_t)(n0 + l15) * CORRK + kb;

    v8f acc[4] = {};
    Frag aC, bC0, bC1, bC2, bC3;
    loadFrag(aC, abase, 0);
    loadFrag(bC0, bbase, 0 * 16 * CORRK);
    loadFrag(bC1, bbase, 1 * 16 * CORRK);
    loadFrag(bC2, bbase, 2 * 16 * CORRK);
    loadFrag(bC3, bbase, 3 * 16 * CORRK);
#pragma unroll
    for (int s = 0; s < 3; ++s) {
        Frag aN, bN0, bN1, bN2, bN3;
        if (s + 1 < 3) {
            const int kc = (s + 1) * 32;
            loadFrag(aN, abase, kc);
            loadFrag(bN0, bbase, 0 * 16 * CORRK + kc);
            loadFrag(bN1, bbase, 1 * 16 * CORRK + kc);
            loadFrag(bN2, bbase, 2 * 16 * CORRK + kc);
            loadFrag(bN3, bbase, 3 * 16 * CORRK + kc);
        }
        acc[0] = __builtin_amdgcn_wmma_f32_16x16x32_bf16(false, aC.f, false, bC0.f, (short)0, acc[0], false, false);
        acc[1] = __builtin_amdgcn_wmma_f32_16x16x32_bf16(false, aC.f, false, bC1.f, (short)0, acc[1], false, false);
        acc[2] = __builtin_amdgcn_wmma_f32_16x16x32_bf16(false, aC.f, false, bC2.f, (short)0, acc[2], false, false);
        acc[3] = __builtin_amdgcn_wmma_f32_16x16x32_bf16(false, aC.f, false, bC3.f, (short)0, acc[3], false, false);
        aC = aN; bC0 = bN0; bC1 = bN1; bC2 = bN2; bC3 = bN3;
    }

    const int mbase = m0 + ((lane >> 4) * 8);
    unsigned short* sbase =
        res + (((size_t)b * H_ + h) * W_ + (w0 + mbase)) * NPL + (n0 + l15);
#pragma unroll
    for (int t = 0; t < 4; ++t) {
        float lsum = 0.f, lsq = 0.f;
#pragma unroll
        for (int g = 0; g < 8; ++g) {
            float x = acc[t][g];
            sbase[g * NPL + t * 16] = f2bf(x);
            lsum += x; lsq += x * x;
        }
        atomicAdd(&sSum[n0 + t * 16 + l15], lsum);
        atomicAdd(&sSq[n0 + t * 16 + l15], lsq);
    }
    flush_stats(sSum, sSq, stats);
}

// ---------------- BN finalize: a = g*rsqrt(var+eps), b = beta - mean*a ----------------
__global__ void bn_finalize_kernel(const float* __restrict__ stats, float* __restrict__ params,
                                   const float* __restrict__ g, const float* __restrict__ beta,
                                   float invN) {
    int c = threadIdx.x;                       // 128 threads
    float mean = stats[c] * invN;
    float var  = stats[NPL + c] * invN - mean * mean;
    float a    = g[c] * rsqrtf(var + 1e-5f);
    params[c]       = a;
    params[NPL + c] = beta[c] - mean * a;
}

// ---------------- apply BN1+ReLU to act1 interior in place ----------------
// halo stays zero (matches reference: conv2 zero-pads AFTER bn+relu)
__global__ void bn_apply_kernel(unsigned short* __restrict__ act1,
                                const float* __restrict__ bn1, int n) {
    int idx = blockIdx.x * blockDim.x + threadIdx.x;
    if (idx >= n) return;                      // n = B*H*W*128
    int ch = idx & (NPL - 1);
    int p  = idx >> 7;
    int w  = p % W_;
    int h  = (p / W_) % H_;
    int b  = p / (H_ * W_);
    size_t a = (((size_t)b * HP + (h + 1)) * WP + (w + 1)) * NPL + ch;
    float x = bf2f(act1[a]);
    act1[a] = f2bf(fmaxf(fmaf(bn1[ch], x, bn1[NPL + ch]), 0.f));
}

// ---------------- conv2: 3x3, 128x9 -> 128, plain WMMA implicit GEMM ----------------
__global__ void conv2_gemm_kernel(const unsigned short* __restrict__ act1,
                                  const unsigned short* __restrict__ w2,
                                  unsigned short* __restrict__ cv2,
                                  float* __restrict__ stats) {
    __shared__ float sSum[NPL], sSq[NPL];
    if (threadIdx.x < NPL) { sSum[threadIdx.x] = 0.f; sSq[threadIdx.x] = 0.f; }
    __syncthreads();

    const int b = blockIdx.z, h = blockIdx.y, w0 = blockIdx.x * 64;
    const int wave = threadIdx.x >> 5, lane = threadIdx.x & 31;
    const int mw = wave & 3, nw = wave >> 2;
    const int m0 = mw * 16, n0 = nw * 64;
    const int l15 = lane & 15, kb = (lane >> 4) * 8;

    const unsigned short* abase =
        act1 + (((size_t)b * HP + h) * WP + (w0 + m0 + l15)) * NPL + kb;
    const unsigned short* bbase = w2 + (size_t)(n0 + l15) * NPL + kb;

    v8f acc[4] = {};
    constexpr int NS = 36;
    Frag aC, bC0, bC1, bC2, bC3;
    loadFrag(aC, abase, a_off_c2(0));
    loadFrag(bC0, bbase, b_off_c2(0, 0));
    loadFrag(bC1, bbase, b_off_c2(0, 1));
    loadFrag(bC2, bbase, b_off_c2(0, 2));
    loadFrag(bC3, bbase, b_off_c2(0, 3));
#pragma unroll
    for (int s = 0; s < NS; ++s) {
        Frag aN, bN0, bN1, bN2, bN3;
        if (s + 1 < NS) {
            loadFrag(aN, abase, a_off_c2(s + 1));
            loadFrag(bN0, bbase, b_off_c2(s + 1, 0));
            loadFrag(bN1, bbase, b_off_c2(s + 1, 1));
            loadFrag(bN2, bbase, b_off_c2(s + 1, 2));
            loadFrag(bN3, bbase, b_off_c2(s + 1, 3));
        }
        acc[0] = __builtin_amdgcn_wmma_f32_16x16x32_bf16(false, aC.f, false, bC0.f, (short)0, acc[0], false, false);
        acc[1] = __builtin_amdgcn_wmma_f32_16x16x32_bf16(false, aC.f, false, bC1.f, (short)0, acc[1], false, false);
        acc[2] = __builtin_amdgcn_wmma_f32_16x16x32_bf16(false, aC.f, false, bC2.f, (short)0, acc[2], false, false);
        acc[3] = __builtin_amdgcn_wmma_f32_16x16x32_bf16(false, aC.f, false, bC3.f, (short)0, acc[3], false, false);
        aC = aN; bC0 = bN0; bC1 = bN1; bC2 = bN2; bC3 = bN3;
    }

    const int mbase = m0 + ((lane >> 4) * 8);
    unsigned short* sbase =
        cv2 + (((size_t)b * H_ + h) * W_ + (w0 + mbase)) * NPL + (n0 + l15);
#pragma unroll
    for (int t = 0; t < 4; ++t) {
        float lsum = 0.f, lsq = 0.f;
#pragma unroll
        for (int g = 0; g < 8; ++g) {
            float x = acc[t][g];
            sbase[g * NPL + t * 16] = f2bf(x);
            lsum += x; lsq += x * x;
        }
        atomicAdd(&sSum[n0 + t * 16 + l15], lsum);
        atomicAdd(&sSq[n0 + t * 16 + l15], lsq);
    }
    flush_stats(sSum, sSq, stats);
}

// ---------------- final: relu(bn2(conv2) + bn_ds(res)) -> NCHW fp32 ----------------
__global__ void final_kernel(const unsigned short* __restrict__ cv2,
                             const unsigned short* __restrict__ res,
                             const float* __restrict__ bn2, const float* __restrict__ bnd,
                             float* __restrict__ out, int n) {
    int idx = blockIdx.x * blockDim.x + threadIdx.x;
    if (idx >= n) return;
    int w  = idx % W_;
    int h  = (idx / W_) % H_;
    int ch = (idx / (W_ * H_)) % NPL;
    int b  = idx / (NPL * H_ * W_);
    size_t pix = ((size_t)b * H_ + h) * W_ + w;
    float x = bf2f(cv2[pix * NPL + ch]);
    float r = bf2f(res[pix * NPL + ch]);
    float y = fmaf(bn2[ch], x, bn2[NPL + ch]) + fmaf(bnd[ch], r, bnd[NPL + ch]);
    out[idx] = fmaxf(y, 0.f);
}

// ---------------- host launch ----------------
extern "C" void kernel_launch(void* const* d_in, const int* in_sizes, int n_in,
                              void* d_out, int out_size, void* d_ws, size_t ws_size,
                              hipStream_t stream) {
    (void)in_sizes; (void)n_in; (void)out_size; (void)ws_size;
    const float* img1    = (const float*)d_in[0];
    const float* img2    = (const float*)d_in[1];
    const float* w_conv1 = (const float*)d_in[2];
    const float* g1      = (const float*)d_in[3];
    const float* b1      = (const float*)d_in[4];
    const float* w_conv2 = (const float*)d_in[5];
    const float* g2      = (const float*)d_in[6];
    const float* b2      = (const float*)d_in[7];
    const float* w_ds    = (const float*)d_in[8];
    const float* g_ds    = (const float*)d_in[9];
    const float* b_ds    = (const float*)d_in[10];
    float* out = (float*)d_out;

    // workspace carve-up (256B aligned)
    char* ws = (char*)d_ws;
    auto carve = [&](size_t bytes) -> char* {
        char* p = ws; ws += (bytes + 255) & ~(size_t)255; return p;
    };
    const size_t P = (size_t)HP * WP;
    unsigned short* corr = (unsigned short*)carve((size_t)B_ * P * CORRK * 2);
    unsigned short* act1 = (unsigned short*)carve((size_t)B_ * P * NPL * 2);
    unsigned short* cv2  = (unsigned short*)carve((size_t)B_ * H_ * W_ * NPL * 2);
    unsigned short* res  = (unsigned short*)carve((size_t)B_ * H_ * W_ * NPL * 2);
    unsigned short* w1p  = (unsigned short*)carve((size_t)9 * NPL * CORRK * 2);
    unsigned short* w2p  = (unsigned short*)carve((size_t)9 * NPL * NPL * 2);
    unsigned short* wdsp = (unsigned short*)carve((size_t)NPL * CORRK * 2);
    float* stats = (float*)carve(3 * 2 * NPL * sizeof(float));
    float* bnp   = (float*)carve(3 * 2 * NPL * sizeof(float));

    const float invN = 1.f / (float)((size_t)B_ * H_ * W_);
    const dim3 gtile(W_ / 64, H_, B_);

    zero_stats_kernel<<<1, 256, 0, stream>>>(stats);
    { int n = 9 * NPL * CORRK; pack_w1_kernel<<<(n + 255) / 256, 256, 0, stream>>>(w_conv1, w1p, n); }
    { int n = 9 * NPL * NPL;   pack_w2_kernel<<<(n + 255) / 256, 256, 0, stream>>>(w_conv2, w2p, n); }
    { int n = NPL * CORRK;     pack_wds_kernel<<<(n + 255) / 256, 256, 0, stream>>>(w_ds, wdsp, n); }
    { int n = B_ * (int)P;     halo_zero_kernel<<<(n + 255) / 256, 256, 0, stream>>>(corr, act1, n); }

    corr_kernel<<<gtile, 64, 0, stream>>>(img1, img2, corr);
    conv1_gemm_kernel<<<gtile, 256, 0, stream>>>(corr, w1p, act1, stats + 0 * 2 * NPL);
    ds_gemm_kernel<<<gtile, 256, 0, stream>>>(corr, wdsp, res, stats + 1 * 2 * NPL);
    bn_finalize_kernel<<<1, NPL, 0, stream>>>(stats + 0 * 2 * NPL, bnp + 0 * 2 * NPL, g1, b1, invN);
    bn_finalize_kernel<<<1, NPL, 0, stream>>>(stats + 1 * 2 * NPL, bnp + 1 * 2 * NPL, g_ds, b_ds, invN);
    { int n = B_ * H_ * W_ * NPL;
      bn_apply_kernel<<<(n + 255) / 256, 256, 0, stream>>>(act1, bnp + 0 * 2 * NPL, n); }
    conv2_gemm_kernel<<<gtile, 256, 0, stream>>>(act1, w2p, cv2, stats + 2 * 2 * NPL);
    bn_finalize_kernel<<<1, NPL, 0, stream>>>(stats + 2 * 2 * NPL, bnp + 2 * 2 * NPL, g2, b2, invN);
    { int n = B_ * NPL * H_ * W_;
      final_kernel<<<(n + 255) / 256, 256, 0, stream>>>(cv2, res, bnp + 2 * 2 * NPL, bnp + 1 * 2 * NPL, out, n); }
}
